// PreCoGenModel_23278722744621
// MI455X (gfx1250) — matrix operand order repaired and checked by reading, not compiled, not run on hardware
//
#include <hip/hip_runtime.h>

// ---------------------------------------------------------------------------
// MI455X (gfx1250) implementation of the PreCo GRU world-model reference.
// All heavy math on v_wmma_f32_16x16x32_f16 (wave32 WMMA), with
// double-buffered LDS staging + register prefetch so global loads overlap
// the WMMA chain and only one workgroup barrier is needed per K-chunk.
// The K loop is phase-split (x then h) so accumulator assignment is
// loop-invariant, and wave-role branches are scalarized via readfirstlane.
// ---------------------------------------------------------------------------

typedef __attribute__((ext_vector_type(16))) _Float16 v16h;
typedef __attribute__((ext_vector_type(8)))  float    v8f;
typedef __attribute__((ext_vector_type(4)))  unsigned int v4u;

#define B_ 4096L
#define T_ 64L
#define S_ 128L
#define A_ 64L
#define L_ 1024L
#define H_ 512L

// ---------------- WMMA helpers (layouts per CDNA5 ISA) ----------

__device__ __forceinline__ v8f wmma16(v16h a, v16h b, v8f c) {
  // D = A(16x32 f16) * B(32x16 f16) + C(16x16 f32)
  return __builtin_amdgcn_wmma_f32_16x16x32_f16(false, a, false, b, (short)0, c,
                                                false, false);
}

// A fragment from a row-major [16][32] f16 LDS tile (ld = 32 halves).
// lane L: row = L&15, g = L>>4; halves 0..7 -> K = g*8..g*8+7,
// halves 8..15 -> K = 16+g*8 .. 16+g*8+7.   (two ds_load_b128)
__device__ __forceinline__ v16h ldsA(const _Float16* p, int lane) {
  int r = lane & 15, g = lane >> 4;
  const _Float16* q = p + r * 32 + g * 8;
  v16h o;
  ((v4u*)&o)[0] = *(const v4u*)(q);
  ((v4u*)&o)[1] = *(const v4u*)(q + 16);
  return o;
}

// B fragment: B(32x16) stored as [n][k] (ld = 32 halves, i.e. W rows verbatim).
// lane L: n = L&15, g = L>>4; halves 0..7 -> K = g*8..g*8+7,
// halves 8..15 -> K = 16+g*8 .. 16+g*8+7.  (two ds_load_b128)
__device__ __forceinline__ v16h ldsB(const _Float16* p, int lane) {
  int n = lane & 15, g = lane >> 4;
  const _Float16* q = p + n * 32 + g * 8;
  v16h o;
  ((v4u*)&o)[0] = *(const v4u*)(q);
  ((v4u*)&o)[1] = *(const v4u*)(q + 16);
  return o;
}

// Store C/D frag (f32, 16x16) into an LDS buffer with row stride 128 floats.
// lane L: n = L&15, g = L>>4; VGPR v -> row m = v + g*8.
__device__ __forceinline__ void stC(float* c, int mb, int nb, v8f acc, int lane) {
  int n = lane & 15, g = lane >> 4;
  float* p = c + (mb + g * 8) * 128 + nb + n;
#pragma unroll
  for (int v = 0; v < 8; ++v) p[v * 128] = acc[v];
}

// ---------------- Fused GRU cell -------------------------------------------
// h_out = GRUCell(x, h_in) for a [64 batch] x [32 latent] tile.
// Strips in Cbuf (f32, 64x128): [0:32)=r pre-act sum, [32:64)=z pre-act sum,
// [64:96)=i_n (x phase only), [96:128)=h_n (h phase only).
// Waves 0-3: r/z strips (both phases). Waves 4-7: n strip — x phase feeds
// (acc0,acc1)=i_n, h phase feeds (acc2,acc3)=h_n (phase-split loops, so the
// accumulator choice is loop-invariant).
__global__ __launch_bounds__(256) void gru_fused(
    const _Float16* __restrict__ x, long x_stride, int Kx,
    const _Float16* __restrict__ h_in,            // [B][L] f16, may be null
    const _Float16* __restrict__ w_ih,            // [3L][Kx] f16
    const _Float16* __restrict__ w_hh,            // [3L][L]  f16
    const float* __restrict__ b_ih, const float* __restrict__ b_hh,
    _Float16* __restrict__ h_out)                 // [B][L] f16
{
  __shared__ _Float16 As[2][64 * 32];
  __shared__ _Float16 Bs[2][96 * 32];
  __shared__ float    Cbuf[64 * 128];
  const int tid = threadIdx.x, lane = tid & 31, wave = tid >> 5;
  const int m0 = blockIdx.x * 64, j0 = blockIdx.y * 32;
  const int mb = (wave & 3) * 16;
  const int arow = tid >> 2, ako = (tid & 3) * 8;
  // wave role is wave-uniform: force it into an SGPR -> uniform s_cbranch,
  // EXEC stays all-ones (required for WMMA).
  const int wave_u = __builtin_amdgcn_readfirstlane(wave);
  const bool low = (wave_u < 4);     // also == (tid < 128) as a wave set
  v8f acc0 = {}, acc1 = {}, acc2 = {}, acc3 = {};

  const int nx = Kx / 32;
  const int nh = h_in ? (int)(L_ / 32) : 0;
  const int niter = nx + nh;

  // global source addresses for global iteration `it`
  auto addrA = [&](int it) -> const _Float16* {
    if (it < nx) return x + (long)(m0 + arow) * x_stride + it * 32 + ako;
    return h_in + (long)(m0 + arow) * L_ + (long)(it - nx) * 32 + ako;
  };
  auto addrB = [&](int it, int c) -> const _Float16* {
    int row = c >> 2, ko = (c & 3) * 8;
    int q = row >> 5, jn = row & 31;
    if (it < nx) return w_ih + (long)(q * L_ + j0 + jn) * Kx + it * 32 + ko;
    return w_hh + (long)(q * L_ + j0 + jn) * L_ + (long)(it - nx) * 32 + ko;
  };
  const int brow0 = tid >> 2, bko0 = (tid & 3) * 8;              // chunk tid
  const int brow1 = (256 + tid) >> 2, bko1 = (tid & 3) * 8;      // chunk 256+tid

  // prefetch iteration 0 into registers
  v4u ra = *(const v4u*)addrA(0);
  v4u rb0 = *(const v4u*)addrB(0, tid);
  v4u rb1;
  if (low) rb1 = *(const v4u*)addrB(0, 256 + tid);

  // ---- phase 1: x @ W_ih^T ----
  for (int it = 0; it < nx; ++it) {
    const int p = it & 1;
    _Float16* asb = As[p];
    _Float16* bsb = Bs[p];
    *(v4u*)(asb + arow * 32 + ako) = ra;
    *(v4u*)(bsb + brow0 * 32 + bko0) = rb0;
    if (low) *(v4u*)(bsb + brow1 * 32 + bko1) = rb1;
    __syncthreads();
    if (it + 1 < niter) {                 // overlap next chunk with WMMAs
      ra = *(const v4u*)addrA(it + 1);
      rb0 = *(const v4u*)addrB(it + 1, tid);
      if (low) rb1 = *(const v4u*)addrB(it + 1, 256 + tid);
    }
    const v16h af = ldsA(asb + mb * 32, lane);
    if (low) {
      v16h bf0 = ldsB(bsb + 0 * 32, lane);
      v16h bf1 = ldsB(bsb + 16 * 32, lane);
      v16h bf2 = ldsB(bsb + 32 * 32, lane);
      v16h bf3 = ldsB(bsb + 48 * 32, lane);
      acc0 = wmma16(af, bf0, acc0);
      acc1 = wmma16(af, bf1, acc1);
      acc2 = wmma16(af, bf2, acc2);
      acc3 = wmma16(af, bf3, acc3);
    } else {
      v16h bf0 = ldsB(bsb + 64 * 32, lane);
      v16h bf1 = ldsB(bsb + 80 * 32, lane);
      acc0 = wmma16(af, bf0, acc0);       // i_n
      acc1 = wmma16(af, bf1, acc1);
    }
    __syncthreads();
  }

  // ---- phase 2: h @ W_hh^T (absent for h==0 start: gh = b_hh) ----
  for (int it = nx; it < niter; ++it) {
    const int p = it & 1;
    _Float16* asb = As[p];
    _Float16* bsb = Bs[p];
    *(v4u*)(asb + arow * 32 + ako) = ra;
    *(v4u*)(bsb + brow0 * 32 + bko0) = rb0;
    if (low) *(v4u*)(bsb + brow1 * 32 + bko1) = rb1;
    __syncthreads();
    if (it + 1 < niter) {
      ra = *(const v4u*)addrA(it + 1);
      rb0 = *(const v4u*)addrB(it + 1, tid);
      if (low) rb1 = *(const v4u*)addrB(it + 1, 256 + tid);
    }
    const v16h af = ldsA(asb + mb * 32, lane);
    if (low) {
      v16h bf0 = ldsB(bsb + 0 * 32, lane);
      v16h bf1 = ldsB(bsb + 16 * 32, lane);
      v16h bf2 = ldsB(bsb + 32 * 32, lane);
      v16h bf3 = ldsB(bsb + 48 * 32, lane);
      acc0 = wmma16(af, bf0, acc0);
      acc1 = wmma16(af, bf1, acc1);
      acc2 = wmma16(af, bf2, acc2);
      acc3 = wmma16(af, bf3, acc3);
    } else {
      v16h bf0 = ldsB(bsb + 64 * 32, lane);
      v16h bf1 = ldsB(bsb + 80 * 32, lane);
      acc2 = wmma16(af, bf0, acc2);       // h_n
      acc3 = wmma16(af, bf1, acc3);
    }
    __syncthreads();
  }

  const int cb = low ? 0 : 64;
  stC(Cbuf, mb, cb +  0, acc0, lane);
  stC(Cbuf, mb, cb + 16, acc1, lane);
  stC(Cbuf, mb, cb + 32, acc2, lane);
  stC(Cbuf, mb, cb + 48, acc3, lane);
  __syncthreads();

  // gate fusion epilogue
  for (int idx = tid; idx < 64 * 32; idx += 256) {
    int m = idx >> 5, j = idx & 31;
    int jg = j0 + j;
    long b = m0 + m;
    float rr  = Cbuf[m * 128 + j]       + b_ih[jg]           + b_hh[jg];
    float zz  = Cbuf[m * 128 + 32 + j]  + b_ih[L_ + jg]      + b_hh[L_ + jg];
    float in_ = Cbuf[m * 128 + 64 + j]  + b_ih[2 * L_ + jg];
    float hn  = Cbuf[m * 128 + 96 + j]  + b_hh[2 * L_ + jg];
    float r = 1.f / (1.f + expf(-rr));
    float z = 1.f / (1.f + expf(-zz));
    float n = tanhf(in_ + r * hn);
    float hp = h_in ? (float)h_in[b * L_ + jg] : 0.f;
    h_out[b * L_ + jg] = (_Float16)((1.f - z) * n + z * hp);
  }
}

// ---------------- Decoder stage 1: relu(h @ fc1_w^T + b) -------------------
__global__ __launch_bounds__(256) void dec1_kernel(
    const _Float16* __restrict__ h,     // [B][1024]
    const _Float16* __restrict__ w,     // [512][1024]
    const float* __restrict__ bias,     // [512]
    _Float16* __restrict__ out)         // [B][512]
{
  __shared__ _Float16 As[2][64 * 32];
  __shared__ _Float16 Bs[2][64 * 32];
  const int tid = threadIdx.x, lane = tid & 31, wave = tid >> 5;
  const int m0 = blockIdx.x * 64, n0 = blockIdx.y * 64;
  const int mb = (wave & 3) * 16, nb = (wave >> 2) * 32;
  const int row = tid >> 2, ko = (tid & 3) * 8;
  v8f acc0 = {}, acc1 = {};
  v4u ra = *(const v4u*)(h + (long)(m0 + row) * L_ + ko);
  v4u rb = *(const v4u*)(w + (long)(n0 + row) * L_ + ko);
  const int niter = (int)(L_ / 32);
  for (int it = 0; it < niter; ++it) {
    const int p = it & 1;
    *(v4u*)(As[p] + row * 32 + ko) = ra;
    *(v4u*)(Bs[p] + row * 32 + ko) = rb;
    __syncthreads();
    if (it + 1 < niter) {
      ra = *(const v4u*)(h + (long)(m0 + row) * L_ + (it + 1) * 32 + ko);
      rb = *(const v4u*)(w + (long)(n0 + row) * L_ + (it + 1) * 32 + ko);
    }
    v16h af = ldsA(As[p] + mb * 32, lane);
    v16h bf0 = ldsB(Bs[p] + nb * 32, lane);
    v16h bf1 = ldsB(Bs[p] + (nb + 16) * 32, lane);
    acc0 = wmma16(af, bf0, acc0);
    acc1 = wmma16(af, bf1, acc1);
    __syncthreads();
  }
  int n = lane & 15, g = lane >> 4;
#pragma unroll
  for (int v = 0; v < 8; ++v) {
    long bb = m0 + mb + g * 8 + v;
    int nn = n0 + nb + n;
    out[bb * H_ + nn]      = (_Float16)fmaxf(acc0[v] + bias[nn], 0.f);
    out[bb * H_ + nn + 16] = (_Float16)fmaxf(acc1[v] + bias[nn + 16], 0.f);
  }
}

// ------- Decoder stage 2 + loss: sum |tanh(d1 @ fco^T + b)*mask - tgt| -----
__global__ __launch_bounds__(256) void dec2_kernel(
    const _Float16* __restrict__ d1,    // [B][512]
    const _Float16* __restrict__ w,     // [128][512]
    const float* __restrict__ bias,     // [128]
    const float* __restrict__ tgt, long tgt_stride,
    const float* __restrict__ mask, int mask_stride, int mask_off,
    float* __restrict__ accum)
{
  __shared__ _Float16 As[2][64 * 32];
  __shared__ _Float16 Bs[2][64 * 32];
  __shared__ float red[256];
  const int tid = threadIdx.x, lane = tid & 31, wave = tid >> 5;
  const int m0 = blockIdx.x * 64, n0 = blockIdx.y * 64;
  const int mb = (wave & 3) * 16, nb = (wave >> 2) * 32;
  const int row = tid >> 2, ko = (tid & 3) * 8;
  v8f acc0 = {}, acc1 = {};
  v4u ra = *(const v4u*)(d1 + (long)(m0 + row) * H_ + ko);
  v4u rb = *(const v4u*)(w + (long)(n0 + row) * H_ + ko);
  const int niter = (int)(H_ / 32);
  for (int it = 0; it < niter; ++it) {
    const int p = it & 1;
    *(v4u*)(As[p] + row * 32 + ko) = ra;
    *(v4u*)(Bs[p] + row * 32 + ko) = rb;
    __syncthreads();
    if (it + 1 < niter) {
      ra = *(const v4u*)(d1 + (long)(m0 + row) * H_ + (it + 1) * 32 + ko);
      rb = *(const v4u*)(w + (long)(n0 + row) * H_ + (it + 1) * 32 + ko);
    }
    v16h af = ldsA(As[p] + mb * 32, lane);
    v16h bf0 = ldsB(Bs[p] + nb * 32, lane);
    v16h bf1 = ldsB(Bs[p] + (nb + 16) * 32, lane);
    acc0 = wmma16(af, bf0, acc0);
    acc1 = wmma16(af, bf1, acc1);
    __syncthreads();
  }
  int n = lane & 15, g = lane >> 4;
  float part = 0.f;
#pragma unroll
  for (int v = 0; v < 8; ++v) {
    long bb = m0 + mb + g * 8 + v;
    float mk = mask[bb * mask_stride + mask_off];
    int nn = n0 + nb + n;
    float o0 = tanhf(acc0[v] + bias[nn]) * mk;
    float o1 = tanhf(acc1[v] + bias[nn + 16]) * mk;
    part += fabsf(o0 - tgt[bb * tgt_stride + nn]);
    part += fabsf(o1 - tgt[bb * tgt_stride + nn + 16]);
  }
  red[tid] = part;
  __syncthreads();
  for (int s = 128; s > 0; s >>= 1) {
    if (tid < s) red[tid] += red[tid + s];
    __syncthreads();
  }
  if (tid == 0) atomicAdd(accum, red[0]);
}

// ---------------- Misc small kernels ---------------------------------------
__global__ void cvt_f16(const float* __restrict__ s, _Float16* __restrict__ d, long n) {
  for (long i = (long)blockIdx.x * blockDim.x + threadIdx.x; i < n;
       i += (long)gridDim.x * blockDim.x)
    d[i] = (_Float16)s[i];
}

__global__ void gather_x0(const float* __restrict__ x, _Float16* __restrict__ d) {
  long i = (long)blockIdx.x * blockDim.x + threadIdx.x;
  if (i < B_ * S_) {
    long b = i >> 7, s = i & 127;
    d[i] = (_Float16)x[b * (T_ * S_) + s];   // obs[0][b][s] = x[b][0][s]
  }
}

__global__ void mask_kernel(const float* __restrict__ src, long rstride, int K,
                            float* __restrict__ m, long rows) {
  long r = (long)blockIdx.x * blockDim.x + threadIdx.x;
  if (r < rows) {
    float s = 0.f;
    for (int k = 0; k < K; ++k) s += fabsf(src[r * rstride + k]);
    m[r] = (s > 0.f) ? 1.f : 0.f;
  }
}

__global__ void zero3(float* a) {
  if (threadIdx.x < 3) a[threadIdx.x] = 0.f;
}

__global__ void finalize(const float* __restrict__ acc, float* __restrict__ out) {
  if (threadIdx.x == 0) {
    float denom = (float)(B_ * S_);
    float corr = acc[0] / denom;
    float single = acc[1] / denom;
    float seq = acc[2] / (63.f * denom) + single;
    out[0] = corr;
    out[1] = single;
    out[2] = seq;
  }
}

// ---------------- Host-side orchestration ----------------------------------
extern "C" void kernel_launch(void* const* d_in, const int* in_sizes, int n_in,
                              void* d_out, int out_size, void* d_ws, size_t ws_size,
                              hipStream_t stream) {
  (void)in_sizes; (void)n_in; (void)out_size; (void)ws_size;
  const float* x        = (const float*)d_in[0];   // [B][T][128]
  const float* a        = (const float*)d_in[1];   // [B][T][64]
  const float* y        = (const float*)d_in[2];   // [B][T][128]
  const float* pred_wih = (const float*)d_in[3];
  const float* pred_whh = (const float*)d_in[4];
  const float* pred_bih = (const float*)d_in[5];
  const float* pred_bhh = (const float*)d_in[6];
  const float* corr_wih = (const float*)d_in[7];
  const float* corr_whh = (const float*)d_in[8];
  const float* corr_bih = (const float*)d_in[9];
  const float* corr_bhh = (const float*)d_in[10];
  const float* fc1_w    = (const float*)d_in[11];
  const float* fc1_b    = (const float*)d_in[12];
  const float* fco_w    = (const float*)d_in[13];
  const float* fco_b    = (const float*)d_in[14];

  char* w8 = (char*)d_ws;
  size_t off = 0;
  auto take = [&](size_t bytes) {
    char* p = w8 + off;
    off = (off + bytes + 255) & ~(size_t)255;
    return p;
  };
  float*    acc  = (float*)take(16);
  _Float16* a16  = (_Float16*)take(B_ * T_ * A_ * 2);
  _Float16* x016 = (_Float16*)take(B_ * S_ * 2);
  _Float16* pwih = (_Float16*)take(3 * L_ * A_ * 2);
  _Float16* pwhh = (_Float16*)take(3 * L_ * L_ * 2);
  _Float16* cwih = (_Float16*)take(3 * L_ * S_ * 2);
  _Float16* cwhh = (_Float16*)take(3 * L_ * L_ * 2);
  _Float16* fc1  = (_Float16*)take(H_ * L_ * 2);
  _Float16* fco  = (_Float16*)take(S_ * H_ * 2);
  _Float16* hA   = (_Float16*)take(B_ * L_ * 2);
  _Float16* hB   = (_Float16*)take(B_ * L_ * 2);
  _Float16* d1b  = (_Float16*)take(B_ * H_ * 2);
  float*    mska = (float*)take(B_ * T_ * 4);
  float*    mskx = (float*)take(B_ * 4);

  // --- one-time precision conversion + masks + accumulator reset ---
  cvt_f16<<<4096, 256, 0, stream>>>(a, a16, B_ * T_ * A_);
  gather_x0<<<(int)((B_ * S_ + 255) / 256), 256, 0, stream>>>(x, x016);
  cvt_f16<<<1024, 256, 0, stream>>>(pred_wih, pwih, 3 * L_ * A_);
  cvt_f16<<<4096, 256, 0, stream>>>(pred_whh, pwhh, 3 * L_ * L_);
  cvt_f16<<<1024, 256, 0, stream>>>(corr_wih, cwih, 3 * L_ * S_);
  cvt_f16<<<4096, 256, 0, stream>>>(corr_whh, cwhh, 3 * L_ * L_);
  cvt_f16<<<2048, 256, 0, stream>>>(fc1_w, fc1, H_ * L_);
  cvt_f16<<<256, 256, 0, stream>>>(fco_w, fco, S_ * H_);
  mask_kernel<<<(int)((B_ * T_ + 255) / 256), 256, 0, stream>>>(a, A_, (int)A_, mska, B_ * T_);
  mask_kernel<<<(int)((B_ + 255) / 256), 256, 0, stream>>>(x, T_ * S_, (int)S_, mskx, B_);
  zero3<<<1, 32, 0, stream>>>(acc);

  dim3 gGru(B_ / 64, L_ / 32);
  dim3 gD1(B_ / 64, H_ / 64);
  dim3 gD2(B_ / 64, S_ / 64);

  // --- corr step: GRU(obs[0], h=0) -> decode, loss vs obs[0] ---
  gru_fused<<<gGru, 256, 0, stream>>>(x016, S_, (int)S_, nullptr, cwih, cwhh,
                                      corr_bih, corr_bhh, hA);
  dec1_kernel<<<gD1, 256, 0, stream>>>(hA, fc1, fc1_b, d1b);
  dec2_kernel<<<gD2, 256, 0, stream>>>(d1b, fco, fco_b, x, T_ * S_, mskx, 1, 0,
                                       acc + 0);

  // --- single step: GRU(act[0], corr_h) -> decode, loss vs new_obs[0] ---
  gru_fused<<<gGru, 256, 0, stream>>>(a16, T_ * A_, (int)A_, hA, pwih, pwhh,
                                      pred_bih, pred_bhh, hB);
  dec1_kernel<<<gD1, 256, 0, stream>>>(hB, fc1, fc1_b, d1b);
  dec2_kernel<<<gD2, 256, 0, stream>>>(d1b, fco, fco_b, y, T_ * S_, mska,
                                       (int)T_, 0, acc + 1);

  // --- seq scan: t = 1..63 ---
  _Float16* cur = hB;
  _Float16* nxt = hA;
  for (int t = 1; t < (int)T_; ++t) {
    gru_fused<<<gGru, 256, 0, stream>>>(a16 + (long)t * A_, T_ * A_, (int)A_,
                                        cur, pwih, pwhh, pred_bih, pred_bhh, nxt);
    dec1_kernel<<<gD1, 256, 0, stream>>>(nxt, fc1, fc1_b, d1b);
    dec2_kernel<<<gD2, 256, 0, stream>>>(d1b, fco, fco_b, y + (long)t * S_,
                                         T_ * S_, mska, (int)T_, t, acc + 2);
    _Float16* tmp = cur; cur = nxt; nxt = tmp;
  }

  finalize<<<1, 32, 0, stream>>>(acc, (float*)d_out);
}
